// LSTMAutoencoder_38216618999993
// MI455X (gfx1250) — matrix operand order repaired and checked by reading, not compiled, not run on hardware
//
#include <hip/hip_runtime.h>

// LSTM autoencoder, persistent kernel: B=256,T=512,D_IN=32,H=128,LAT=64
#define B_SZ 256
#define T_SZ 512
#define DIN  32
#define H_SZ 128
#define LATD 64
#define G4H  512

// LDS strides (in halves / floats), padded for bank-conflict avoidance
#define WSTR 168   // gate weight K-stride (K up to 160)
#define HSTR 136   // h row stride (128 + pad)
#define XSTR 40    // x_t row stride (32 + pad)
#define LSTR 136   // dec_Wl K-stride
#define ZSTR 520   // zg row stride (floats)

#define HBUF (16 * HSTR)
#define XBUF (16 * XSTR)

#define SMEM_BYTES (((G4H*WSTR + 2*HBUF + 2*XBUF + 32*LSTR) * 2) + \
                    ((16*ZSTR + 16*LATD + G4H + 32) * 4))

typedef __attribute__((ext_vector_type(16))) _Float16 v16h;
typedef __attribute__((ext_vector_type(8)))  _Float16 v8h;
typedef __attribute__((ext_vector_type(8)))  float    v8f;

__device__ __forceinline__ v8f wmma_f16(v16h a, v16h b, v8f c) {
  return __builtin_amdgcn_wmma_f32_16x16x32_f16(false, a, false, b,
                                                (short)0, c, false, false);
}

__device__ __forceinline__ float fast_rcp(float x) {
#if __has_builtin(__builtin_amdgcn_rcpf)
  return __builtin_amdgcn_rcpf(x);
#else
  return 1.0f / x;
#endif
}
__device__ __forceinline__ float sigm(float x) {
  return fast_rcp(1.0f + __expf(-x));
}
__device__ __forceinline__ float fast_tanh(float x) {
#if __has_builtin(__builtin_amdgcn_tanhf)
  return __builtin_amdgcn_tanhf(x);
#else
  float e = __expf(2.0f * x);
  return (e - 1.0f) * fast_rcp(e + 1.0f);
#endif
}

// A-fragment (16-bit A 16x32): elements 0..7 = K base, 8..15 = K base+8
__device__ __forceinline__ v16h ldsA(const _Float16* base) {
  v8h lo = *(const v8h*)(base);
  v8h hi = *(const v8h*)(base + 8);
  v16h a;
#pragma unroll
  for (int i = 0; i < 8; ++i) { a[i] = lo[i]; a[i + 8] = hi[i]; }
  return a;
}

__global__ __launch_bounds__(256, 1) void lstm_ae(
    const float* __restrict__ x,
    const float* __restrict__ eWih, const float* __restrict__ eWhh,
    const float* __restrict__ ebih, const float* __restrict__ ebhh,
    const float* __restrict__ eWl,  const float* __restrict__ ebl,
    const float* __restrict__ dWih, const float* __restrict__ dWhh,
    const float* __restrict__ dbih, const float* __restrict__ dbhh,
    const float* __restrict__ dWl,  const float* __restrict__ dbl,
    float* __restrict__ out_enc, float* __restrict__ out_dec)
{
  extern __shared__ char smem[];
  _Float16* Wg = (_Float16*)smem;          // [512][WSTR] gate weights (K-major per column)
  _Float16* Hb = Wg + G4H * WSTR;          // [2][16][HSTR]  hidden state f16 (double buffer)
  _Float16* Xb = Hb + 2 * HBUF;            // [2][16][XSTR]  x_t f16 (double buffer)
  _Float16* Wl = Xb + 2 * XBUF;            // [32][LSTR]  dec_Wl
  float* ZG    = (float*)(Wl + 32 * LSTR); // [16][ZSTR]  z@Wih.T + biases
  float* Zf    = ZG + 16 * ZSTR;           // [16][64]    encoded f32
  float* BiasE = Zf + 16 * LATD;           // [512]       enc bih+bhh
  float* BiasO = BiasE + G4H;              // [32]        dec_bl

  const int tid  = threadIdx.x;
  const int wave = tid >> 5;
  const int lane = tid & 31;
  const int ln   = lane & 15;
  const int lg   = lane >> 4;
  const int row0 = blockIdx.x * 16;       // 16 batch rows per workgroup
  const int hoff = ln * HSTR + lg * 16;   // A-frag base within an H buffer
  const int xoff = ln * XSTR + lg * 16;

  // ---------- one-time setup ----------
  // encoder combined weights: k<128 = Whh.T column, k in [128,160) = Wih.T
  for (int i = tid; i < G4H * 160; i += 256) {
    int n = i / 160, k = i - n * 160;
    float v = (k < H_SZ) ? eWhh[n * H_SZ + k] : eWih[n * DIN + (k - H_SZ)];
    Wg[n * WSTR + k] = (_Float16)v;
  }
  for (int i = tid; i < G4H; i += 256) BiasE[i] = ebih[i] + ebhh[i];
  for (int i = tid; i < 32 * H_SZ; i += 256) {
    int n = i >> 7, k = i & 127;
    Wl[n * LSTR + k] = (_Float16)dWl[n * H_SZ + k];
  }
  if (tid < 32) BiasO[tid] = dbl[tid];
  for (int i = tid; i < 2 * HBUF; i += 256) Hb[i] = (_Float16)0.f;
  {
    int e0 = tid, e1 = tid + 256;       // 512 elements of x tile [16][32]
    float v0 = x[((row0 + (e0 >> 5)) * T_SZ + 0) * DIN + (e0 & 31)];
    float v1 = x[((row0 + (e1 >> 5)) * T_SZ + 0) * DIN + (e1 & 31)];
    Xb[(e0 >> 5) * XSTR + (e0 & 31)] = (_Float16)v0;   // buffer 0
    Xb[(e1 >> 5) * XSTR + (e1 & 31)] = (_Float16)v1;
  }
  __syncthreads();

  // hoisted loop invariants: recurrent-weight B fragments + gate biases
  v16h bw[4][4];
#pragma unroll
  for (int j = 0; j < 4; ++j) {
    int n0 = (j * 8 + wave) * 16;
#pragma unroll
    for (int kb = 0; kb < 4; ++kb)
      bw[j][kb] = *(const v16h*)&Wg[(n0 + ln) * WSTR + kb * 32 + lg * 16];
  }
  const float bi = BiasE[0 * 128 + wave * 16 + ln];
  const float bf = BiasE[1 * 128 + wave * 16 + ln];
  const float bg = BiasE[2 * 128 + wave * 16 + ln];
  const float bo = BiasE[3 * 128 + wave * 16 + ln];

  v8f c = {};   // cell state fragment: rows r+8*lg, cols wave*16+ln

  // ===================== encoder scan (1 barrier/step) =====================
  for (int t = 0; t < T_SZ; ++t) {
    const _Float16* rH = Hb + (t & 1) * HBUF;          // h_{t-1}
    const _Float16* rX = Xb + (t & 1) * XBUF;          // x_t
    _Float16* wH = Hb + ((t + 1) & 1) * HBUF;          // h_t
    _Float16* wX = Xb + ((t + 1) & 1) * XBUF;          // x_{t+1}

    v16h a[5];
#pragma unroll
    for (int kb = 0; kb < 4; ++kb)
      a[kb] = ldsA(rH + hoff + kb * 32);
    a[4] = ldsA(rX + xoff);

    float px0 = 0.f, px1 = 0.f;     // prefetch x_{t+1} during WMMA phase
    if (t + 1 < T_SZ) {
      int e0 = tid, e1 = tid + 256;
      px0 = x[((row0 + (e0 >> 5)) * T_SZ + (t + 1)) * DIN + (e0 & 31)];
      px1 = x[((row0 + (e1 >> 5)) * T_SZ + (t + 1)) * DIN + (e1 & 31)];
    }

    v8f acc[4];
#pragma unroll
    for (int j = 0; j < 4; ++j) {   // j = gate (i,f,g,o), tile N = j*8 + wave
      int n0 = (j * 8 + wave) * 16;
      v16h bx = *(const v16h*)&Wg[(n0 + ln) * WSTR + 4 * 32 + lg * 16];
      v8f z = {};
      z = wmma_f16(a[4], bx, z);                       // x_t contribution
#pragma unroll
      for (int kb = 0; kb < 4; ++kb)
        z = wmma_f16(a[kb], bw[j][kb], z);             // h_{t-1} contribution
      acc[j] = z;
    }

#pragma unroll
    for (int r = 0; r < 8; ++r) {
      float iv = sigm(acc[0][r] + bi);
      float fv = sigm(acc[1][r] + bf);
      float gv = fast_tanh(acc[2][r] + bg);
      float ov = sigm(acc[3][r] + bo);
      float cv = fv * c[r] + iv * gv;
      c[r] = cv;
      float hv = ov * fast_tanh(cv);
      wH[(r + 8 * lg) * HSTR + wave * 16 + ln] = (_Float16)hv;
    }
    if (t + 1 < T_SZ) {
      int e0 = tid, e1 = tid + 256;
      wX[(e0 >> 5) * XSTR + (e0 & 31)] = (_Float16)px0;
      wX[(e1 >> 5) * XSTR + (e1 & 31)] = (_Float16)px1;
    }
    __syncthreads();   // h_t / x_{t+1} visible; prior-buffer reads long done
  }

  // h_T lives in buffer 0 (written at t=511).
  // ===================== encoded = h_T @ eWl.T + ebl (tiny; VALU) ==========
  for (int i = tid; i < 16 * LATD; i += 256) {
    int m = i >> 6, n = i & 63;
    float s = ebl[n];
    for (int k = 0; k < H_SZ; ++k)
      s += (float)Hb[m * HSTR + k] * eWl[n * H_SZ + k];
    Zf[m * LATD + n] = s;
    out_enc[(row0 + m) * LATD + n] = s;
  }
  __syncthreads();   // Zf ready; encoder done with Wg and Hb

  // decoder recurrent weights overwrite Wg; zg = z@dWih.T + biases (time-const)
  for (int i = tid; i < G4H * H_SZ; i += 256) {
    int n = i >> 7, k = i & 127;
    Wg[n * WSTR + k] = (_Float16)dWhh[n * H_SZ + k];
  }
  for (int i = tid; i < 16 * G4H; i += 256) {
    int m = i >> 9, n = i & 511;
    float s = dbih[n] + dbhh[n];
    for (int j = 0; j < LATD; ++j)
      s += Zf[m * LATD + j] * dWih[n * LATD + j];
    ZG[m * ZSTR + n] = s;
  }
  for (int i = tid; i < HBUF; i += 256) Hb[i] = (_Float16)0.f;   // h_{-1}=0 in buf0
  __syncthreads();

  // hoisted decoder invariants: Whh fragments + zg C-init fragments
#pragma unroll
  for (int j = 0; j < 4; ++j) {
    int n0 = (j * 8 + wave) * 16;
#pragma unroll
    for (int kb = 0; kb < 4; ++kb)
      bw[j][kb] = *(const v16h*)&Wg[(n0 + ln) * WSTR + kb * 32 + lg * 16];
  }
  v8f zg[4];
#pragma unroll
  for (int j = 0; j < 4; ++j) {
    int n0 = (j * 8 + wave) * 16;
#pragma unroll
    for (int r = 0; r < 8; ++r)
      zg[j][r] = ZG[(r + 8 * lg) * ZSTR + n0 + ln];
  }
  const float bO = BiasO[wave * 16 + ln];
#pragma unroll
  for (int r = 0; r < 8; ++r) c[r] = 0.f;

  // ===================== decoder scan (1 barrier/step) =====================
  for (int t = 0; t < T_SZ; ++t) {
    const _Float16* rH = Hb + (t & 1) * HBUF;          // h_{t-1}
    _Float16* wH = Hb + ((t + 1) & 1) * HBUF;          // h_t

    v16h a[4];
#pragma unroll
    for (int kb = 0; kb < 4; ++kb)
      a[kb] = ldsA(rH + hoff + kb * 32);

    // fused output projection for the PREVIOUS step: decoded[:,t-1,:] uses
    // exactly these A fragments (h_{t-1}); waves 0-1 own the 2 N-tiles.
    if (wave < 2 && t > 0) {
      v8f oacc = {};
#pragma unroll
      for (int kb = 0; kb < 4; ++kb) {
        v16h b = *(const v16h*)&Wl[(wave * 16 + ln) * LSTR + kb * 32 + lg * 16];
        oacc = wmma_f16(a[kb], b, oacc);
      }
#pragma unroll
      for (int r = 0; r < 8; ++r) {
        int m = r + 8 * lg;
        out_dec[((row0 + m) * T_SZ + (t - 1)) * DIN + wave * 16 + ln] =
            oacc[r] + bO;
      }
    }

    v8f acc[4];
#pragma unroll
    for (int j = 0; j < 4; ++j) {
      v8f z = zg[j];                 // C-input = precomputed z@Wih.T + biases
#pragma unroll
      for (int kb = 0; kb < 4; ++kb)
        z = wmma_f16(a[kb], bw[j][kb], z);
      acc[j] = z;
    }

#pragma unroll
    for (int r = 0; r < 8; ++r) {
      float iv = sigm(acc[0][r]);
      float fv = sigm(acc[1][r]);
      float gv = fast_tanh(acc[2][r]);
      float ov = sigm(acc[3][r]);
      float cv = fv * c[r] + iv * gv;
      c[r] = cv;
      float hv = ov * fast_tanh(cv);
      wH[(r + 8 * lg) * HSTR + wave * 16 + ln] = (_Float16)hv;
    }
    __syncthreads();   // h_t visible
  }

  // epilogue: output projection for t = 511 (h_511 is in buffer 0)
  if (wave < 2) {
    v8f oacc = {};
#pragma unroll
    for (int kb = 0; kb < 4; ++kb) {
      v16h a2 = ldsA(Hb + hoff + kb * 32);
      v16h b  = *(const v16h*)&Wl[(wave * 16 + ln) * LSTR + kb * 32 + lg * 16];
      oacc = wmma_f16(a2, b, oacc);
    }
#pragma unroll
    for (int r = 0; r < 8; ++r) {
      int m = r + 8 * lg;
      out_dec[((row0 + m) * T_SZ + (T_SZ - 1)) * DIN + wave * 16 + ln] =
          oacc[r] + bO;
    }
  }
}

extern "C" void kernel_launch(void* const* d_in, const int* in_sizes, int n_in,
                              void* d_out, int out_size, void* d_ws, size_t ws_size,
                              hipStream_t stream) {
  const float* x    = (const float*)d_in[0];
  const float* eWih = (const float*)d_in[1];
  const float* eWhh = (const float*)d_in[2];
  const float* ebih = (const float*)d_in[3];
  const float* ebhh = (const float*)d_in[4];
  const float* eWl  = (const float*)d_in[5];
  const float* ebl  = (const float*)d_in[6];
  const float* dWih = (const float*)d_in[7];
  const float* dWhh = (const float*)d_in[8];
  const float* dbih = (const float*)d_in[9];
  const float* dbhh = (const float*)d_in[10];
  const float* dWl  = (const float*)d_in[11];
  const float* dbl  = (const float*)d_in[12];
  float* out = (float*)d_out;

  (void)hipFuncSetAttribute((const void*)lstm_ae,
                            hipFuncAttributeMaxDynamicSharedMemorySize,
                            SMEM_BYTES);
  lstm_ae<<<dim3(B_SZ / 16), dim3(256), SMEM_BYTES, stream>>>(
      x, eWih, eWhh, ebih, ebhh, eWl, ebl,
      dWih, dWhh, dbih, dbhh, dWl, dbl,
      out, out + B_SZ * LATD);
}
